// NunchakuOminiFluxTransformerBlocks_65051574665871
// MI455X (gfx1250) — matrix operand order repaired and checked by reading, not compiled, not run on hardware
//
#include <hip/hip_runtime.h>
#include <stdint.h>

// ---------------------------------------------------------------------------
// Types for CDNA5 WMMA (wave32): v_wmma_f32_16x16x32_bf16
// ---------------------------------------------------------------------------
typedef __attribute__((ext_vector_type(16))) __bf16 v16bf;
typedef __attribute__((ext_vector_type(8)))  float  v8f;

__device__ __forceinline__ v8f vzero8() {
  v8f z = {0.f, 0.f, 0.f, 0.f, 0.f, 0.f, 0.f, 0.f};
  return z;
}

__device__ __forceinline__ v8f wmma_bf16(v16bf a, v16bf b, v8f c) {
  // 8 args: (neg_a, A, neg_b, B, c_mod, C, reuse_a, reuse_b)
  return __builtin_amdgcn_wmma_f32_16x16x32_bf16(false, a, false, b, (short)0, c,
                                                 false, false);
}

__device__ __forceinline__ float gelu_tanh(float x) {
  const float k0 = 0.7978845608028654f;  // sqrt(2/pi)
  float x3 = x * x * x;
  return 0.5f * x * (1.0f + tanhf(k0 * (x + 0.044715f * x3)));
}

#define D3072 3072
#define STOT  2048   // 512 txt + 1024 img + 512 cond
#define NHEAD 24
#define DHEAD 128

// ===========================================================================
// 1) Fused SiLU + modulation GEMV:  out[j] = silu(t) @ W[:, j] + b[j]
//    Optionally a second activation vector sharing the same weight read.
//    W is (3072, 18432) row-major.
// ===========================================================================
__global__ __launch_bounds__(256) void mod_gemv_kernel(
    const float* __restrict__ t1, const float* __restrict__ t2,
    const float* __restrict__ W, const float* __restrict__ b,
    float* __restrict__ o1, float* __restrict__ o2) {
  __shared__ float a1[D3072];
  __shared__ float a2[D3072];
  int tid = threadIdx.x;
  for (int i = tid; i < D3072; i += 256) {
    float v = t1[i];
    a1[i] = v / (1.0f + __expf(-v));
    if (t2) {
      float u = t2[i];
      a2[i] = u / (1.0f + __expf(-u));
    }
  }
  __syncthreads();
  int j = blockIdx.x * 256 + tid;  // 0..18431
  float acc1 = 0.f, acc2 = 0.f;
  for (int k = 0; k < D3072; ++k) {
    float w = W[(size_t)k * 18432 + j];
    acc1 += a1[k] * w;
    if (t2) acc2 += a2[k] * w;
  }
  o1[j] = acc1 + b[j];
  if (t2) o2[j] = acc2 + b[j];
}

// ===========================================================================
// 2) Pre-attention LayerNorm + modulate for the 3 concatenated streams.
//    rows [0,512): txt(enc), [512,1536): img(hidden), [1536,2048): cond
//    out = ln(x) * (1 + mod[1]) + mod[0]
// ===========================================================================
__global__ __launch_bounds__(256) void preln_kernel(
    const float* __restrict__ txt, const float* __restrict__ img,
    const float* __restrict__ cnd, const float* __restrict__ mt,
    const float* __restrict__ mi, const float* __restrict__ mc,
    float* __restrict__ out) {
  __shared__ float red[256];
  int r = blockIdx.x, tid = threadIdx.x;
  const float* x;
  const float* m;
  if (r < 512)        { x = txt + (size_t)r * D3072;          m = mt; }
  else if (r < 1536)  { x = img + (size_t)(r - 512) * D3072;  m = mi; }
  else                { x = cnd + (size_t)(r - 1536) * D3072; m = mc; }

  float s = 0.f, s2 = 0.f;
#pragma unroll
  for (int i = 0; i < 12; ++i) {
    float v = x[tid + 256 * i];
    s += v; s2 += v * v;
  }
  red[tid] = s; __syncthreads();
  for (int o = 128; o > 0; o >>= 1) { if (tid < o) red[tid] += red[tid + o]; __syncthreads(); }
  float mean = red[0] * (1.f / 3072.f);
  __syncthreads();
  red[tid] = s2; __syncthreads();
  for (int o = 128; o > 0; o >>= 1) { if (tid < o) red[tid] += red[tid + o]; __syncthreads(); }
  float var = red[0] * (1.f / 3072.f) - mean * mean;
  float rstd = rsqrtf(var + 1e-6f);

  float* orow = out + (size_t)r * D3072;
#pragma unroll
  for (int i = 0; i < 12; ++i) {
    int d = tid + 256 * i;
    float v = (x[d] - mean) * rstd;
    orow[d] = v * (1.0f + m[D3072 + d]) + m[d];
  }
}

// ===========================================================================
// 3) Post LayerNorm + modulate (single contiguous buffer, txt/img mods).
//    out = ln(x) * (1 + mod[sc]) + mod[so]
// ===========================================================================
__global__ __launch_bounds__(256) void postln_kernel(
    const float* __restrict__ x_, const float* __restrict__ mt,
    const float* __restrict__ mi, float* __restrict__ out, int so, int sc) {
  __shared__ float red[256];
  int r = blockIdx.x, tid = threadIdx.x;
  const float* x = x_ + (size_t)r * D3072;
  const float* m = (r < 512) ? mt : mi;

  float s = 0.f, s2 = 0.f;
#pragma unroll
  for (int i = 0; i < 12; ++i) {
    float v = x[tid + 256 * i];
    s += v; s2 += v * v;
  }
  red[tid] = s; __syncthreads();
  for (int o = 128; o > 0; o >>= 1) { if (tid < o) red[tid] += red[tid + o]; __syncthreads(); }
  float mean = red[0] * (1.f / 3072.f);
  __syncthreads();
  red[tid] = s2; __syncthreads();
  for (int o = 128; o > 0; o >>= 1) { if (tid < o) red[tid] += red[tid + o]; __syncthreads(); }
  float var = red[0] * (1.f / 3072.f) - mean * mean;
  float rstd = rsqrtf(var + 1e-6f);

  float* orow = out + (size_t)r * D3072;
#pragma unroll
  for (int i = 0; i < 12; ++i) {
    int d = tid + 256 * i;
    float v = (x[d] - mean) * rstd;
    orow[d] = v * (1.0f + m[sc * D3072 + d]) + m[so * D3072 + d];
  }
}

// ===========================================================================
// 4) Gated residual: out[r] = x[r] + mod[gateOff] * y[r]   (txt/img select)
//    xa supplies rows [0,512), xb supplies rows [512,1536).
// ===========================================================================
__global__ __launch_bounds__(256) void axpy_gate_kernel(
    const float* __restrict__ xa, const float* __restrict__ xb,
    const float* __restrict__ mt, const float* __restrict__ mi,
    const float* __restrict__ y, float* __restrict__ out, int gateOff) {
  int r = blockIdx.x, tid = threadIdx.x;
  const float* x = (r < 512) ? xa + (size_t)r * D3072 : xb + (size_t)(r - 512) * D3072;
  const float* m = ((r < 512) ? mt : mi) + (size_t)gateOff * D3072;
  const float* yr = y + (size_t)r * D3072;
  float* orow = out + (size_t)r * D3072;
#pragma unroll
  for (int i = 0; i < 12; ++i) {
    int d = tid + 256 * i;
    orow[d] = x[d] + m[d] * yr[d];
  }
}

// ===========================================================================
// 5) Generic tiled WMMA GEMM:  C = act(A @ B + bias)
//    A: M x K fp32 row-major, B: K x N fp32 row-major, C: M x N fp32.
//    Tile 128x128, K-step 32, double-buffered LDS (ping-pong) with the next
//    K-tile's global loads held in registers while WMMAs drain the current
//    buffer. fp32 -> bf16 conversion happens at the LDS-store stage.
//    ACT: 0 = none, 1 = gelu(tanh)
// ===========================================================================
#define BM 128
#define BN 128
#define BK 32
#define APAD 8
#define BPAD 8

template <int ACT>
__global__ __launch_bounds__(256) void gemm_wmma_kernel(
    const float* __restrict__ A, const float* __restrict__ B,
    const float* __restrict__ bias, float* __restrict__ C,
    int M, int N, int K) {
  __shared__ __bf16 As[2][BM][BK + APAD];
  __shared__ __bf16 Bs[2][BK][BN + BPAD];

  int tid = threadIdx.x;
  int bn = blockIdx.x, bm = blockIdx.y;
  int wave = tid >> 5, lane = tid & 31;
  int wm = wave & 1, wn = wave >> 1;   // 2 x 4 wave grid
  int lm = lane & 15, hi = lane >> 4;

  v8f acc[4][2];
#pragma unroll
  for (int i = 0; i < 4; ++i)
#pragma unroll
    for (int j = 0; j < 2; ++j) acc[i][j] = vzero8();

  const float* Ab = A + (size_t)bm * BM * K;
  const float* Bb = B + (size_t)bn * BN;

  // per-thread staging coordinates
  const int ar = tid >> 3;          // A row stride-32 passes
  const int ac = (tid & 7) * 4;     // A col (float4)
  const int br = tid >> 5;          // B row stride-8 passes
  const int bc = (tid & 31) * 4;    // B col (float4)

  float4 ra[4], rb[4];
  auto gload = [&](int k0) {
#pragma unroll
    for (int p = 0; p < 4; ++p)
      ra[p] = *(const float4*)(Ab + (size_t)(ar + p * 32) * K + k0 + ac);
#pragma unroll
    for (int p = 0; p < 4; ++p)
      rb[p] = *(const float4*)(Bb + (size_t)(k0 + br + p * 8) * N + bc);
  };
  auto stage = [&](int buf) {
#pragma unroll
    for (int p = 0; p < 4; ++p) {
      int r = ar + p * 32;
      As[buf][r][ac + 0] = (__bf16)ra[p].x;
      As[buf][r][ac + 1] = (__bf16)ra[p].y;
      As[buf][r][ac + 2] = (__bf16)ra[p].z;
      As[buf][r][ac + 3] = (__bf16)ra[p].w;
    }
#pragma unroll
    for (int p = 0; p < 4; ++p) {
      int r = br + p * 8;
      Bs[buf][r][bc + 0] = (__bf16)rb[p].x;
      Bs[buf][r][bc + 1] = (__bf16)rb[p].y;
      Bs[buf][r][bc + 2] = (__bf16)rb[p].z;
      Bs[buf][r][bc + 3] = (__bf16)rb[p].w;
    }
  };

  // prologue: tile 0 into buffer 0
  gload(0);
  stage(0);
  __syncthreads();

  const int nt = K / BK;
  for (int t = 0; t < nt; ++t) {
    int cur = t & 1;
    if (t + 1 < nt) {
      gload((t + 1) * BK);  // next tile's global loads in flight
      if (t + 2 < nt) {     // prefetch tile t+2 -> global_prefetch_b8
        __builtin_prefetch(Ab + (size_t)ar * K + (t + 2) * BK, 0, 1);
        __builtin_prefetch(Bb + (size_t)((t + 2) * BK + br) * N + bc, 0, 1);
      }
    }

    // ---- A fragments (16x32 per m-tile) from LDS[cur] ----
    v16bf af[4];
#pragma unroll
    for (int i = 0; i < 4; ++i) {
      int row = wm * 64 + i * 16 + lm;
#pragma unroll
      for (int v = 0; v < 8; ++v) {
        int kk = ((v >> 2) << 4) + hi * 8 + ((v & 3) << 1);
        af[i][2 * v + 0] = As[cur][row][kk + 0];
        af[i][2 * v + 1] = As[cur][row][kk + 1];
      }
    }
    // ---- B fragments (32x16 per n-tile) from LDS[cur] ----
    v16bf bfr[2];
#pragma unroll
    for (int j = 0; j < 2; ++j) {
      int col = wn * 32 + j * 16 + lm;
#pragma unroll
      for (int v = 0; v < 8; ++v) {
        int kk = ((v >> 2) << 4) + hi * 8 + ((v & 3) << 1);
        bfr[j][2 * v + 0] = Bs[cur][kk + 0][col];
        bfr[j][2 * v + 1] = Bs[cur][kk + 1][col];
      }
    }
#pragma unroll
    for (int i = 0; i < 4; ++i)
#pragma unroll
      for (int j = 0; j < 2; ++j) acc[i][j] = wmma_bf16(af[i], bfr[j], acc[i][j]);

    if (t + 1 < nt) stage(cur ^ 1);
    __syncthreads();
  }

  // ---- epilogue: bias (+ activation), fp32 store ----
#pragma unroll
  for (int i = 0; i < 4; ++i) {
#pragma unroll
    for (int j = 0; j < 2; ++j) {
      int col = bn * BN + wn * 32 + j * 16 + lm;
      float bcol = bias[col];
#pragma unroll
      for (int r = 0; r < 8; ++r) {
        int row = bm * BM + wm * 64 + i * 16 + r + hi * 8;
        float v = acc[i][j][r] + bcol;
        if (ACT == 1) v = gelu_tanh(v);
        C[(size_t)row * N + col] = v;
      }
    }
  }
}

// ===========================================================================
// 6) RMS-norm + RoPE + head scatter.  One wave per (token s, head h).
//    Reads qkv [2048][9216], writes bf16 Q/K/V as [H][2048][128].
// ===========================================================================
__global__ __launch_bounds__(256) void qk_rope_kernel(
    const float* __restrict__ qkv, const float* __restrict__ tqn,
    const float* __restrict__ tkn, const float* __restrict__ iqn,
    const float* __restrict__ ikn, const float* __restrict__ rot_main,
    const float* __restrict__ rot_cond, __bf16* __restrict__ Qb,
    __bf16* __restrict__ Kb, __bf16* __restrict__ Vb) {
  int g = blockIdx.x * 8 + (threadIdx.x >> 5);  // (s,h) pair index
  int lane = threadIdx.x & 31;
  int s = g / NHEAD;
  int h = g % NHEAD;
  const float* qn = (s < 512) ? tqn : iqn;
  const float* kn = (s < 512) ? tkn : ikn;
  const float* rp = (s < 1536) ? rot_main + (size_t)s * 128
                               : rot_cond + (size_t)(s - 1536) * 128;
  int d = lane * 4;
  const float* base = qkv + (size_t)s * 9216 + h * DHEAD + d;
  float4 q = *(const float4*)(base);
  float4 k = *(const float4*)(base + 3072);
  float4 v = *(const float4*)(base + 6144);

  float sq = q.x * q.x + q.y * q.y + q.z * q.z + q.w * q.w;
  float sk = k.x * k.x + k.y * k.y + k.z * k.z + k.w * k.w;
#pragma unroll
  for (int m = 1; m < 32; m <<= 1) {
    sq += __shfl_xor(sq, m, 32);
    sk += __shfl_xor(sk, m, 32);
  }
  float rq = rsqrtf(sq * (1.f / 128.f) + 1e-6f);
  float rk = rsqrtf(sk * (1.f / 128.f) + 1e-6f);

  float q0 = q.x * rq * qn[d + 0], q1 = q.y * rq * qn[d + 1];
  float q2 = q.z * rq * qn[d + 2], q3 = q.w * rq * qn[d + 3];
  float k0 = k.x * rk * kn[d + 0], k1 = k.y * rk * kn[d + 1];
  float k2 = k.z * rk * kn[d + 2], k3 = k.w * rk * kn[d + 3];

  // rope: rp layout per pair p: [2p]=cos, [2p+1]=sin; d=4*lane -> rp[d..d+3]
  float c0 = rp[d + 0], s0 = rp[d + 1];
  float c1 = rp[d + 2], s1 = rp[d + 3];
  float qo0 = q0 * c0 - q1 * s0, qo1 = q0 * s0 + q1 * c0;
  float qo2 = q2 * c1 - q3 * s1, qo3 = q2 * s1 + q3 * c1;
  float ko0 = k0 * c0 - k1 * s0, ko1 = k0 * s0 + k1 * c0;
  float ko2 = k2 * c1 - k3 * s1, ko3 = k2 * s1 + k3 * c1;

  size_t ob = ((size_t)h * STOT + s) * DHEAD + d;
  Qb[ob + 0] = (__bf16)qo0; Qb[ob + 1] = (__bf16)qo1;
  Qb[ob + 2] = (__bf16)qo2; Qb[ob + 3] = (__bf16)qo3;
  Kb[ob + 0] = (__bf16)ko0; Kb[ob + 1] = (__bf16)ko1;
  Kb[ob + 2] = (__bf16)ko2; Kb[ob + 3] = (__bf16)ko3;
  Vb[ob + 0] = (__bf16)v.x; Vb[ob + 1] = (__bf16)v.y;
  Vb[ob + 2] = (__bf16)v.z; Vb[ob + 3] = (__bf16)v.w;
}

// ===========================================================================
// 7) Flash attention with WMMA bf16.  Block = (head, 128 q rows), 8 waves,
//    each wave owns 16 q rows with online softmax. K/V tiles (32 keys, bf16,
//    no conversion needed) are staged to LDS with the CDNA5 async path
//    (GLOBAL_LOAD_ASYNC_TO_LDS_B128, ASYNCcnt-tracked). P is staged per-wave
//    in LDS to re-enter as an A fragment. Output merged into [S][H*DH] fp32.
// ===========================================================================
__global__ __launch_bounds__(256) void attn_kernel(
    const __bf16* __restrict__ Qb, const __bf16* __restrict__ Kb,
    const __bf16* __restrict__ Vb, float* __restrict__ O) {
  __shared__ __bf16 Ks[32][DHEAD + 8];
  __shared__ __bf16 Vs[32][DHEAD + 8];
  __shared__ __bf16 Ps[8][16][32 + 8];

  int h = blockIdx.x >> 4;
  int qblk = blockIdx.x & 15;
  int tid = threadIdx.x;
  int wave = tid >> 5, lane = tid & 31;
  int lm = lane & 15, hi = lane >> 4;
  int qrow0 = qblk * 128 + wave * 16;

  const __bf16* Qh = Qb + (size_t)h * STOT * DHEAD;
  const __bf16* Kh = Kb + (size_t)h * STOT * DHEAD;
  const __bf16* Vh = Vb + (size_t)h * STOT * DHEAD;

  // per-thread staging coordinates (16 threads/row, 8 bf16 = 16B each)
  const int sr = tid >> 4;
  const int sc = (tid & 15) * 8;

  // Q fragments: 4 chunks of 16x32 along head dim
  v16bf aq[4];
#pragma unroll
  for (int c = 0; c < 4; ++c) {
    const __bf16* qr = Qh + (size_t)(qrow0 + lm) * DHEAD;
#pragma unroll
    for (int v = 0; v < 8; ++v) {
      int kk = c * 32 + ((v >> 2) << 4) + hi * 8 + ((v & 3) << 1);
      aq[c][2 * v + 0] = qr[kk + 0];
      aq[c][2 * v + 1] = qr[kk + 1];
    }
  }

  float mrow[8], lrow[8];
  v8f accO[8];
#pragma unroll
  for (int r = 0; r < 8; ++r) { mrow[r] = -3.0e38f; lrow[r] = 0.f; }
#pragma unroll
  for (int t = 0; t < 8; ++t) accO[t] = vzero8();

  const float scale = 0.08838834764831845f;  // 1/sqrt(128)

  for (int kt = 0; kt < STOT; kt += 32) {
    // stage K/V tiles (32 x 128 bf16) via async global->LDS DMA
#pragma unroll
    for (int p = 0; p < 2; ++p) {
      int r = sr + p * 16;
      uint32_t dK = (uint32_t)(uintptr_t)&Ks[r][sc];
      uint32_t dV = (uint32_t)(uintptr_t)&Vs[r][sc];
      const __bf16* gK = Kh + (size_t)(kt + r) * DHEAD + sc;
      const __bf16* gV = Vh + (size_t)(kt + r) * DHEAD + sc;
      asm volatile("global_load_async_to_lds_b128 %0, %1, off"
                   :: "v"(dK), "v"(gK) : "memory");
      asm volatile("global_load_async_to_lds_b128 %0, %1, off"
                   :: "v"(dV), "v"(gV) : "memory");
    }
    asm volatile("s_wait_asynccnt 0x0" ::: "memory");
    __syncthreads();

    // S = Q @ K^T  (16 x 32 scores per wave)
    v8f accS[2];
    accS[0] = vzero8(); accS[1] = vzero8();
#pragma unroll
    for (int j = 0; j < 2; ++j) {
#pragma unroll
      for (int c = 0; c < 4; ++c) {
        v16bf bk;
#pragma unroll
        for (int v = 0; v < 8; ++v) {
          int kd = c * 32 + ((v >> 2) << 4) + hi * 8 + ((v & 3) << 1);
          bk[2 * v + 0] = Ks[j * 16 + lm][kd + 0];
          bk[2 * v + 1] = Ks[j * 16 + lm][kd + 1];
        }
        accS[j] = wmma_bf16(aq[c], bk, accS[j]);
      }
    }

    // online softmax (rows r+8*hi, replicated across the 16-lane half)
    float alpha[8];
#pragma unroll
    for (int r = 0; r < 8; ++r) {
      float s0 = accS[0][r] * scale;
      float s1 = accS[1][r] * scale;
      float mx = fmaxf(s0, s1);
#pragma unroll
      for (int m = 1; m < 16; m <<= 1) mx = fmaxf(mx, __shfl_xor(mx, m, 16));
      float mn = fmaxf(mrow[r], mx);
      alpha[r] = __expf(mrow[r] - mn);
      mrow[r] = mn;
      float p0 = __expf(s0 - mn);
      float p1 = __expf(s1 - mn);
      float ts = p0 + p1;
#pragma unroll
      for (int m = 1; m < 16; m <<= 1) ts += __shfl_xor(ts, m, 16);
      lrow[r] = lrow[r] * alpha[r] + ts;
      Ps[wave][r + 8 * hi][lm] = (__bf16)p0;
      Ps[wave][r + 8 * hi][16 + lm] = (__bf16)p1;
    }
#pragma unroll
    for (int t = 0; t < 8; ++t)
#pragma unroll
      for (int r = 0; r < 8; ++r) accO[t][r] *= alpha[r];
    __syncthreads();

    // P (16x32, A-layout from LDS) @ V (32x128)
    v16bf ap;
#pragma unroll
    for (int v = 0; v < 8; ++v) {
      int kk = ((v >> 2) << 4) + hi * 8 + ((v & 3) << 1);
      ap[2 * v + 0] = Ps[wave][lm][kk + 0];
      ap[2 * v + 1] = Ps[wave][lm][kk + 1];
    }
#pragma unroll
    for (int t = 0; t < 8; ++t) {
      v16bf bv;
#pragma unroll
      for (int v = 0; v < 8; ++v) {
        int key = ((v >> 2) << 4) + hi * 8 + ((v & 3) << 1);
        bv[2 * v + 0] = Vs[key + 0][t * 16 + lm];
        bv[2 * v + 1] = Vs[key + 1][t * 16 + lm];
      }
      accO[t] = wmma_bf16(ap, bv, accO[t]);
    }
    __syncthreads();
  }

  // merged write: O[s][h*128 + d]
#pragma unroll
  for (int t = 0; t < 8; ++t) {
#pragma unroll
    for (int r = 0; r < 8; ++r) {
      int qr = qrow0 + r + 8 * hi;
      int d = t * 16 + lm;
      O[(size_t)qr * (NHEAD * DHEAD) + h * DHEAD + d] = accO[t][r] / lrow[r];
    }
  }
}

// ===========================================================================
// Host-side orchestration
// ===========================================================================
extern "C" void kernel_launch(void* const* d_in, const int* in_sizes, int n_in,
                              void* d_out, int out_size, void* d_ws, size_t ws_size,
                              hipStream_t stream) {
  (void)in_sizes; (void)n_in; (void)out_size; (void)ws_size;

  const float* hidden   = (const float*)d_in[0];   // (1,1024,3072)
  const float* cond_hid = (const float*)d_in[1];   // (1,512,3072)
  const float* temb     = (const float*)d_in[2];   // (1,3072)
  const float* cond_temb= (const float*)d_in[3];   // (1,3072)
  const float* enc      = (const float*)d_in[4];   // (1,512,3072)
  const float* rot_main = (const float*)d_in[5];   // (1,1,1536,64,1,2)
  const float* rot_cond = (const float*)d_in[6];   // (1,1,512,64,1,2)

  const float* img_mod_w = (const float*)d_in[7];
  const float* img_mod_b = (const float*)d_in[8];
  const float* img_qkv_w = (const float*)d_in[9];
  const float* img_qkv_b = (const float*)d_in[10];
  const float* img_qnorm = (const float*)d_in[11];
  const float* img_knorm = (const float*)d_in[12];
  const float* img_proj_w = (const float*)d_in[13];
  const float* img_proj_b = (const float*)d_in[14];
  const float* img_mlp_w1 = (const float*)d_in[15];
  const float* img_mlp_b1 = (const float*)d_in[16];
  const float* img_mlp_w2 = (const float*)d_in[17];
  const float* img_mlp_b2 = (const float*)d_in[18];

  const float* txt_mod_w = (const float*)d_in[19];
  const float* txt_mod_b = (const float*)d_in[20];
  const float* txt_qkv_w = (const float*)d_in[21];
  const float* txt_qkv_b = (const float*)d_in[22];
  const float* txt_qnorm = (const float*)d_in[23];
  const float* txt_knorm = (const float*)d_in[24];
  const float* txt_proj_w = (const float*)d_in[25];
  const float* txt_proj_b = (const float*)d_in[26];
  const float* txt_mlp_w1 = (const float*)d_in[27];
  const float* txt_mlp_b1 = (const float*)d_in[28];
  const float* txt_mlp_w2 = (const float*)d_in[29];
  const float* txt_mlp_b2 = (const float*)d_in[30];

  // -------- workspace layout (with aliasing where lifetimes allow) --------
  char* p = (char*)d_ws;
  auto alloc = [&](size_t bytes) -> char* {
    char* r = p;
    p += (bytes + 255) & ~(size_t)255;
    return r;
  };
  float* img_mod = (float*)alloc(18432 * sizeof(float));
  float* cond_mod = (float*)alloc(18432 * sizeof(float));
  float* txt_mod = (float*)alloc(18432 * sizeof(float));
  float* xhat    = (float*)alloc((size_t)STOT * D3072 * sizeof(float));   // aliased: attn_out
  float* qkv     = (float*)alloc((size_t)STOT * 9216 * sizeof(float));    // aliased: hbuf (1536x12288)
  __bf16* Qb = (__bf16*)alloc((size_t)NHEAD * STOT * DHEAD * sizeof(__bf16));
  __bf16* Kb = (__bf16*)alloc((size_t)NHEAD * STOT * DHEAD * sizeof(__bf16));
  __bf16* Vb = (__bf16*)alloc((size_t)NHEAD * STOT * DHEAD * sizeof(__bf16));
  float* proj = (float*)alloc((size_t)1536 * D3072 * sizeof(float));      // aliased: mlp2 out
  float* xres = (float*)alloc((size_t)1536 * D3072 * sizeof(float));
  float* ybuf = (float*)alloc((size_t)1536 * D3072 * sizeof(float));
  float* attn_out = xhat;                // xhat dead after QKV GEMMs
  float* hbuf = qkv;                     // qkv dead after qk_rope; same size
  float* m2o = proj;                     // proj dead after first residual
  float* out = (float*)d_out;            // [txt 512 | img 1024] x 3072

  // 1) modulation vectors
  mod_gemv_kernel<<<72, 256, 0, stream>>>(temb, cond_temb, img_mod_w, img_mod_b,
                                          img_mod, cond_mod);
  mod_gemv_kernel<<<72, 256, 0, stream>>>(temb, nullptr, txt_mod_w, txt_mod_b,
                                          txt_mod, nullptr);

  // 2) pre-attention LN + modulate (all 3 streams -> xhat rows 0..2047)
  preln_kernel<<<STOT, 256, 0, stream>>>(enc, hidden, cond_hid, txt_mod, img_mod,
                                         cond_mod, xhat);

  // 3) QKV projections (txt rows 0..511; img+cond rows 512..2047 share weights)
  gemm_wmma_kernel<0><<<dim3(9216 / BN, 512 / BM), 256, 0, stream>>>(
      xhat, txt_qkv_w, txt_qkv_b, qkv, 512, 9216, D3072);
  gemm_wmma_kernel<0><<<dim3(9216 / BN, 1536 / BM), 256, 0, stream>>>(
      xhat + (size_t)512 * D3072, img_qkv_w, img_qkv_b,
      qkv + (size_t)512 * 9216, 1536, 9216, D3072);

  // 4) RMS norm + RoPE + head scatter -> bf16 Q/K/V [H][2048][128]
  qk_rope_kernel<<<(STOT * NHEAD) / 8, 256, 0, stream>>>(
      qkv, txt_qnorm, txt_knorm, img_qnorm, img_knorm, rot_main, rot_cond,
      Qb, Kb, Vb);

  // 5) attention (flash, WMMA, async K/V staging) -> merged [2048][3072]
  attn_kernel<<<NHEAD * 16, 256, 0, stream>>>(Qb, Kb, Vb, attn_out);

  // 6) output projections
  gemm_wmma_kernel<0><<<dim3(D3072 / BN, 512 / BM), 256, 0, stream>>>(
      attn_out, txt_proj_w, txt_proj_b, proj, 512, D3072, D3072);
  gemm_wmma_kernel<0><<<dim3(D3072 / BN, 1024 / BM), 256, 0, stream>>>(
      attn_out + (size_t)512 * D3072, img_proj_w, img_proj_b,
      proj + (size_t)512 * D3072, 1024, D3072, D3072);

  // 7) x = x_in + gate_msa * proj
  axpy_gate_kernel<<<1536, 256, 0, stream>>>(enc, hidden, txt_mod, img_mod, proj,
                                             xres, 2);

  // 8) y = ln(x) * (1 + scale_mlp) + shift_mlp
  postln_kernel<<<1536, 256, 0, stream>>>(xres, txt_mod, img_mod, ybuf, 3, 4);

  // 9) MLP up + gelu
  gemm_wmma_kernel<1><<<dim3(12288 / BN, 512 / BM), 256, 0, stream>>>(
      ybuf, txt_mlp_w1, txt_mlp_b1, hbuf, 512, 12288, D3072);
  gemm_wmma_kernel<1><<<dim3(12288 / BN, 1024 / BM), 256, 0, stream>>>(
      ybuf + (size_t)512 * D3072, img_mlp_w1, img_mlp_b1,
      hbuf + (size_t)512 * 12288, 1024, 12288, D3072);

  // 10) MLP down
  gemm_wmma_kernel<0><<<dim3(D3072 / BN, 512 / BM), 256, 0, stream>>>(
      hbuf, txt_mlp_w2, txt_mlp_b2, m2o, 512, D3072, 12288);
  gemm_wmma_kernel<0><<<dim3(D3072 / BN, 1024 / BM), 256, 0, stream>>>(
      hbuf + (size_t)512 * 12288, img_mlp_w2, img_mlp_b2,
      m2o + (size_t)512 * D3072, 1024, D3072, 12288);

  // 11) out = x + gate_mlp * mlp_out  (rows already in [txt | img] order)
  axpy_gate_kernel<<<1536, 256, 0, stream>>>(xres, xres + (size_t)512 * D3072,
                                             txt_mod, img_mod, m2o, out, 5);
}
